// NeuralSynth_5909874999416
// MI455X (gfx1250) — matrix operand order repaired and checked by reading, not compiled, not run on hardware
//
#include <hip/hip_runtime.h>

typedef __attribute__((ext_vector_type(16))) __bf16 v16bf;
typedef __attribute__((ext_vector_type(8)))  __bf16 v8bf;
typedef __attribute__((ext_vector_type(8)))  float  v8f;

#define B_SZ   8
#define T_SZ   400
#define H_SZ   512
#define P_SZ   64
#define M_TOT  (B_SZ * T_SZ)      // 3200
#define UPS    160
#define N_SMP  (T_SZ * UPS)       // 64000

__device__ inline v8f zero8() {
  v8f z;
#pragma unroll
  for (int i = 0; i < 8; ++i) z[i] = 0.f;
  return z;
}

// ---- WMMA fragment loaders (CDNA5 16x16x32 bf16 layouts) ----
// A (16x32, MxK): lane m = lane&15, kbase = (lane>>4)*8; elems 0..7 -> K=kbase+i,
// elems 8..15 -> K=16+kbase+(i-8).
__device__ inline v16bf load_a_global(const __bf16* __restrict__ X, int ldx,
                                      int m0, int kc, int lane) {
  int m  = m0 + (lane & 15);
  int kb = kc + ((lane >> 4) << 3);
  const __bf16* p = X + (size_t)m * ldx + kb;
  v8bf lo = *(const v8bf*)p;
  v8bf hi = *(const v8bf*)(p + 16);
  v16bf a;
#pragma unroll
  for (int i = 0; i < 8; ++i) { a[i] = lo[i]; a[i + 8] = hi[i]; }
  return a;
}
// B (32x16, KxN) for Y = X @ W^T with W row-major (N,K): lane n = lane&15,
// kbase = (lane>>4)*8; elems 0..7 -> K = kbase+i, elems 8..15 -> K = 16+kbase+(i-8).
__device__ inline v16bf load_b_w(const __bf16* __restrict__ W, int K,
                                 int n0, int kc, int lane) {
  int n  = n0 + (lane & 15);
  int kb = kc + ((lane >> 4) << 3);
  const __bf16* p = W + (size_t)n * K + kb;
  v8bf lo = *(const v8bf*)p;
  v8bf hi = *(const v8bf*)(p + 16);
  v16bf b;
#pragma unroll
  for (int i = 0; i < 8; ++i) { b[i] = lo[i]; b[i + 8] = hi[i]; }
  return b;
}

// ---- f32 -> bf16 convert ----
__global__ void cvt_bf16_kernel(const float* __restrict__ s, __bf16* __restrict__ d, int n) {
  int i = blockIdx.x * blockDim.x + threadIdx.x;
  if (i < n) d[i] = (__bf16)s[i];
}

// ---- first MLP layer (fan_in = 1) + LayerNorm + ReLU ----
__global__ void mlp_first_kernel(const float* __restrict__ x, const float* __restrict__ W,
                                 const float* __restrict__ b, const float* __restrict__ g,
                                 const float* __restrict__ be, __bf16* __restrict__ Y,
                                 int ldy, int ycol0) {
  __shared__ float red[256], red2[256];
  int m = blockIdx.x, t = threadIdx.x;
  float xv = x[m];
  float h0 = xv * W[t] + b[t];
  float h1 = xv * W[t + 256] + b[t + 256];
  red[t] = h0 + h1;
  red2[t] = h0 * h0 + h1 * h1;
  __syncthreads();
  for (int off = 128; off > 0; off >>= 1) {
    if (t < off) { red[t] += red[t + off]; red2[t] += red2[t + off]; }
    __syncthreads();
  }
  float mu  = red[0] * (1.f / 512.f);
  float var = red2[0] * (1.f / 512.f) - mu * mu;
  float rs  = rsqrtf(var + 1e-5f);
  float v0 = fmaxf((h0 - mu) * rs * g[t] + be[t], 0.f);
  float v1 = fmaxf((h1 - mu) * rs * g[t + 256] + be[t + 256], 0.f);
  Y[(size_t)m * ldy + ycol0 + t]       = (__bf16)v0;
  Y[(size_t)m * ldy + ycol0 + t + 256] = (__bf16)v1;
}

// ---- generic WMMA GEMM: Y = X(M,K)bf16 @ W(N,K)^T + bias, optional LN+ReLU ----
// 256 threads = 8 waves; one 16-row M block per blockIdx.x; 16*8*TPW cols per blockIdx.y.
template <int TPW>
__global__ void __launch_bounds__(256) gemm_kernel(
    const __bf16* __restrict__ X, int ldx, int K,
    const __bf16* __restrict__ W, const float* __restrict__ bias,
    const float* __restrict__ g, const float* __restrict__ be,
    __bf16* __restrict__ Ybf, int ldyb, int ycol0,
    float* __restrict__ Yf, int ldyf, int Ntotal, int relu) {
  __shared__ float sY[16][520];
  __shared__ float sred[2][16][16];
  __shared__ float smu[16], srs[16];
  const int lane = threadIdx.x & 31;
  const int wave = threadIdx.x >> 5;
  const int m0   = blockIdx.x << 4;
  const int col0 = blockIdx.y * (16 * 8 * TPW);
  const int ntiles = min((Ntotal - col0) >> 4, 8 * TPW);

  v8f acc[TPW];
#pragma unroll
  for (int i = 0; i < TPW; ++i) acc[i] = zero8();

  for (int kc = 0; kc < K; kc += 32) {
    v16bf a = load_a_global(X, ldx, m0, kc, lane);
#pragma unroll
    for (int i = 0; i < TPW; ++i) {
      int t = wave * TPW + i;
      if (t < ntiles) {
        v16bf bfr = load_b_w(W, K, col0 + (t << 4), kc, lane);
        acc[i] = __builtin_amdgcn_wmma_f32_16x16x32_bf16(
            false, a, false, bfr, (short)0, acc[i], false, false);
      }
    }
  }
  const bool ln = (g != nullptr);
#pragma unroll
  for (int i = 0; i < TPW; ++i) {
    int t = wave * TPW + i;
    if (t >= ntiles) continue;
    int n = col0 + (t << 4) + (lane & 15);
#pragma unroll
    for (int v = 0; v < 8; ++v) {
      int mr = ((lane >> 4) << 3) + v;  // 0..15
      float val = acc[i][v] + bias[n];
      if (ln) {
        sY[mr][n - col0] = val;
      } else {
        if (relu) val = fmaxf(val, 0.f);
        size_t m = (size_t)(m0 + mr);
        if (Yf)  Yf[m * ldyf + n] = val;
        if (Ybf) Ybf[m * ldyb + ycol0 + n] = (__bf16)val;
      }
    }
  }
  if (!ln) return;
  __syncthreads();
  {  // LN stats over 512 cols, 16 rows (assumes Ntotal==512, gridDim.y==1)
    int r = threadIdx.x >> 4, seg = threadIdx.x & 15;
    float s = 0.f, q = 0.f;
#pragma unroll
    for (int j = 0; j < 32; ++j) { float v = sY[r][seg * 32 + j]; s += v; q += v * v; }
    sred[0][r][seg] = s; sred[1][r][seg] = q;
  }
  __syncthreads();
  if (threadIdx.x < 16) {
    int r = threadIdx.x;
    float s = 0.f, q = 0.f;
#pragma unroll
    for (int j = 0; j < 16; ++j) { s += sred[0][r][j]; q += sred[1][r][j]; }
    float mu  = s * (1.f / 512.f);
    float var = q * (1.f / 512.f) - mu * mu;
    smu[r] = mu; srs[r] = rsqrtf(var + 1e-5f);
  }
  __syncthreads();
  for (int e = threadIdx.x; e < 16 * 512; e += 256) {
    int r = e >> 9, n = e & 511;
    float val = (sY[r][n] - smu[r]) * srs[r] * g[n] + be[n];
    if (relu) val = fmaxf(val, 0.f);
    size_t m = (size_t)(m0 + r);
    if (Ybf) Ybf[m * ldyb + ycol0 + n] = (__bf16)val;
    if (Yf)  Yf[m * ldyf + ycol0 + n] = val;
  }
}

// ---- persistent GRU: one workgroup, 16 waves, h kept in LDS, WMMA per step ----
__global__ void __launch_bounds__(512, 1) gru_kernel(
    const float* __restrict__ gi_all,   // (B,T,1536) rows b*T+t
    const __bf16* __restrict__ Whh,     // (1536,512) row-major bf16
    const float* __restrict__ bhh,      // 1536
    __bf16* __restrict__ ys) {          // (B,T,512) rows b*T+t
  __shared__ __align__(16) __bf16 sh[16][512];   // h bf16 (rows 8..15 zero)
  __shared__ float shf[8][512];                  // h f32
  __shared__ float sgh[8][1544];                 // gh = h @ Whh^T
  const int lane = threadIdx.x & 31, wave = threadIdx.x >> 5;
  for (int e = threadIdx.x; e < 16 * 512; e += 512) sh[e >> 9][e & 511] = (__bf16)0.f;
  for (int e = threadIdx.x; e < 8 * 512; e += 512) shf[e >> 9][e & 511] = 0.f;
  __syncthreads();

  for (int t = 0; t < T_SZ; ++t) {
    v8f acc[6];
#pragma unroll
    for (int i = 0; i < 6; ++i) acc[i] = zero8();
    for (int kc = 0; kc < 512; kc += 32) {
      int m = lane & 15, kb = kc + ((lane >> 4) << 3);
      v8bf alo = *(const v8bf*)&sh[m][kb];
      v8bf ahi = *(const v8bf*)&sh[m][kb + 16];
      v16bf a;
#pragma unroll
      for (int i = 0; i < 8; ++i) { a[i] = alo[i]; a[i + 8] = ahi[i]; }
#pragma unroll
      for (int i = 0; i < 6; ++i) {
        int n0 = (wave * 6 + i) << 4;
        v16bf bfr = load_b_w(Whh, 512, n0, kc, lane);
        acc[i] = __builtin_amdgcn_wmma_f32_16x16x32_bf16(
            false, a, false, bfr, (short)0, acc[i], false, false);
      }
    }
#pragma unroll
    for (int i = 0; i < 6; ++i) {
      int n = ((wave * 6 + i) << 4) + (lane & 15);
#pragma unroll
      for (int v = 0; v < 8; ++v) {
        int mr = ((lane >> 4) << 3) + v;
        if (mr < 8) sgh[mr][n] = acc[i][v];
      }
    }
    __syncthreads();
    for (int e = threadIdx.x; e < 8 * 512; e += 512) {
      int b = e >> 9, j = e & 511;
      const float* gi = gi_all + ((size_t)b * T_SZ + t) * 1536;
      float gr = sgh[b][j] + bhh[j];
      float gz = sgh[b][512 + j] + bhh[512 + j];
      float gn = sgh[b][1024 + j] + bhh[1024 + j];
      float r  = 1.f / (1.f + __expf(-(gi[j] + gr)));
      float z  = 1.f / (1.f + __expf(-(gi[512 + j] + gz)));
      float nn = tanhf(gi[1024 + j] + r * gn);
      float hnew = (1.f - z) * nn + z * shf[b][j];
      shf[b][j] = hnew;
      sh[b][j]  = (__bf16)hnew;
      ys[((size_t)b * T_SZ + t) * 512 + j] = (__bf16)hnew;
    }
    __syncthreads();
  }
}

// ---- sigmoid + row-normalize over P=64 ----
__global__ void alpha_kernel(const float* __restrict__ raw, float* __restrict__ alpha) {
  __shared__ float s[64];
  int m = blockIdx.x, t = threadIdx.x;
  float v = 1.f / (1.f + __expf(-raw[(size_t)m * 64 + t]));
  s[t] = v;
  __syncthreads();
  for (int off = 32; off > 0; off >>= 1) {
    if (t < off) s[t] += s[t + off];
    __syncthreads();
  }
  alpha[(size_t)m * 64 + t] = v / s[0];
}

// ---- per-batch prefix sum of upsampled f0 -> phase ----
__global__ void phi_kernel(const float* __restrict__ f0, float* __restrict__ phi) {
  __shared__ float sc[256];
  __shared__ float scarry;
  int b = blockIdx.x, t = threadIdx.x;
  const float* fb = f0 + (size_t)b * T_SZ;
  float f00 = fb[0] * (1.f / 16000.f);
  if (t == 0) scarry = 0.f;
  __syncthreads();
  for (int c = 0; c < N_SMP; c += 256) {
    int n = c + t;
    float pos = (n + 0.5f) * (1.f / (float)UPS) - 0.5f;
    pos = fminf(fmaxf(pos, 0.f), (float)(T_SZ - 1));
    int i0 = (int)pos;
    int i1 = min(i0 + 1, T_SZ - 1);
    float w = pos - (float)i0;
    float v = (fb[i0] * (1.f - w) + fb[i1] * w) * (1.f / 16000.f);
    sc[t] = v;
    __syncthreads();
    for (int off = 1; off < 256; off <<= 1) {
      float tv = (t >= off) ? sc[t - off] : 0.f;
      __syncthreads();
      sc[t] += tv;
      __syncthreads();
    }
    float cum = scarry + sc[t];
    phi[(size_t)b * N_SMP + n] = 6.283185307179586f * (cum - f00);
    __syncthreads();
    if (t == 0) scarry += sc[255];
    __syncthreads();
  }
}

// ---- additive oscillator bank ----
__global__ void synth_kernel(const float* __restrict__ lo, const float* __restrict__ alpha,
                             const float* __restrict__ phi, float* __restrict__ y) {
  int idx = blockIdx.x * blockDim.x + threadIdx.x;
  if (idx >= B_SZ * N_SMP) return;
  int b = idx / N_SMP, n = idx - b * N_SMP;
  float pos = (n + 0.5f) * (1.f / (float)UPS) - 0.5f;
  pos = fminf(fmaxf(pos, 0.f), (float)(T_SZ - 1));
  int i0 = (int)pos;
  int i1 = min(i0 + 1, T_SZ - 1);
  float w = pos - (float)i0;
  const float* lb = lo + (size_t)b * T_SZ;
  float los = lb[i0] * (1.f - w) + lb[i1] * w;
  const float4* a0 = (const float4*)(alpha + ((size_t)b * T_SZ + i0) * 64);
  const float4* a1 = (const float4*)(alpha + ((size_t)b * T_SZ + i1) * 64);
  float ph = phi[idx];
  float sum = 0.f;
#pragma unroll
  for (int q = 0; q < 16; ++q) {
    float4 u = a0[q], v = a1[q];
    float k0 = (float)(4 * q + 1);
    sum += (u.x + (v.x - u.x) * w) * __sinf(ph * k0);
    sum += (u.y + (v.y - u.y) * w) * __sinf(ph * (k0 + 1.f));
    sum += (u.z + (v.z - u.z) * w) * __sinf(ph * (k0 + 2.f));
    sum += (u.w + (v.w - u.w) * w) * __sinf(ph * (k0 + 3.f));
  }
  y[idx] = los * sum;
}

extern "C" void kernel_launch(void* const* d_in, const int* in_sizes, int n_in,
                              void* d_out, int out_size, void* d_ws, size_t ws_size,
                              hipStream_t stream) {
  if (n_in < 44) return;
  auto in = [&](int i) { return (const float*)d_in[i]; };
  const float* f0 = in(0);
  const float* lo = in(1);

  // chain 0 = f0_mlp, 1 = lo_mlp, 2 = fi_mlp
  const float *mW[3][3], *mB[3][3], *mG[3][3], *mBe[3][3];
  const float *Wih, *Whh, *bih, *bhh, *denseW, *denseB;
  bool sorted = (in_sizes[2] == P_SZ * H_SZ);  // dense.W first under sorted-key flatten
  if (sorted) {
    denseW = in(2); denseB = in(3);
    int chains[3] = {0, 2, 1};                 // f0_mlp, fi_mlp, lo_mlp at 4,16,32
    int bases[3]  = {4, 16, 32};
    for (int c = 0; c < 3; ++c)
      for (int l = 0; l < 3; ++l) {            // key order: W, b, be, g
        int p = bases[c] + 4 * l;
        mW[chains[c]][l] = in(p); mB[chains[c]][l] = in(p + 1);
        mBe[chains[c]][l] = in(p + 2); mG[chains[c]][l] = in(p + 3);
      }
    Whh = in(28); Wih = in(29); bhh = in(30); bih = in(31);
  } else {                                     // insertion order
    int chains[3] = {0, 1, 2};                 // f0_mlp, lo_mlp, fi_mlp at 2,14,30
    int bases[3]  = {2, 14, 30};
    for (int c = 0; c < 3; ++c)
      for (int l = 0; l < 3; ++l) {            // key order: W, b, g, be
        int p = bases[c] + 4 * l;
        mW[chains[c]][l] = in(p); mB[chains[c]][l] = in(p + 1);
        mG[chains[c]][l] = in(p + 2); mBe[chains[c]][l] = in(p + 3);
      }
    Wih = in(26); Whh = in(27); bih = in(28); bhh = in(29);
    denseW = in(42); denseB = in(43);
  }

  // ---- workspace carve-up ----
  char* wsp = (char*)d_ws;
  size_t off = 0;
  auto alloc = [&](size_t bytes) {
    void* p = wsp + off;
    off = (off + bytes + 255) & ~(size_t)255;
    return p;
  };
  __bf16* wMlp[3][2];  // bf16 copies of the 512x512 weights (layers 1,2 of f0/lo; fi uses 3)
  __bf16* wFi[3];
  for (int l = 0; l < 2; ++l) wMlp[0][l] = (__bf16*)alloc(H_SZ * H_SZ * 2);
  for (int l = 0; l < 2; ++l) wMlp[1][l] = (__bf16*)alloc(H_SZ * H_SZ * 2);
  for (int l = 0; l < 3; ++l) wFi[l] = (__bf16*)alloc(H_SZ * H_SZ * 2);
  __bf16* wIh    = (__bf16*)alloc((size_t)3 * H_SZ * 2 * H_SZ * 2);  // 1536x1024
  __bf16* wHh    = (__bf16*)alloc((size_t)3 * H_SZ * H_SZ * 2);      // 1536x512
  __bf16* wDen   = (__bf16*)alloc((size_t)P_SZ * H_SZ * 2);          // 64x512
  __bf16* bufA   = (__bf16*)alloc((size_t)M_TOT * H_SZ * 2);
  __bf16* bufB   = (__bf16*)alloc((size_t)M_TOT * H_SZ * 2);
  __bf16* xcat   = (__bf16*)alloc((size_t)M_TOT * 2 * H_SZ * 2);     // 3200x1024
  float*  gi     = (float*)alloc((size_t)M_TOT * 3 * H_SZ * 4);      // 3200x1536
  float*  araw   = (float*)alloc((size_t)M_TOT * P_SZ * 4);
  float*  alpha  = (float*)alloc((size_t)M_TOT * P_SZ * 4);
  float*  phi    = (float*)alloc((size_t)B_SZ * N_SMP * 4);
  (void)ws_size; (void)out_size;
  float* y = (float*)d_out;

  auto cvt = [&](const float* s, __bf16* d, int n) {
    cvt_bf16_kernel<<<(n + 255) / 256, 256, 0, stream>>>(s, d, n);
  };
  cvt(mW[0][1], wMlp[0][0], H_SZ * H_SZ);
  cvt(mW[0][2], wMlp[0][1], H_SZ * H_SZ);
  cvt(mW[1][1], wMlp[1][0], H_SZ * H_SZ);
  cvt(mW[1][2], wMlp[1][1], H_SZ * H_SZ);
  for (int l = 0; l < 3; ++l) cvt(mW[2][l], wFi[l], H_SZ * H_SZ);
  cvt(Wih, wIh, 3 * H_SZ * 2 * H_SZ);
  cvt(Whh, wHh, 3 * H_SZ * H_SZ);
  cvt(denseW, wDen, P_SZ * H_SZ);

  const int MB = M_TOT / 16;  // 200
  auto gemm_ln = [&](const __bf16* X, int ldx, const __bf16* W, const float* bias,
                     const float* g, const float* be, __bf16* Ybf, int ldyb, int ycol0) {
    gemm_kernel<4><<<dim3(MB, 1), 256, 0, stream>>>(
        X, ldx, H_SZ, W, bias, g, be, Ybf, ldyb, ycol0, nullptr, 0, H_SZ, 1);
  };

  // f0 chain -> xcat[:, 0:512]
  mlp_first_kernel<<<M_TOT, 256, 0, stream>>>(f0, mW[0][0], mB[0][0], mG[0][0], mBe[0][0],
                                              bufA, H_SZ, 0);
  gemm_ln(bufA, H_SZ, wMlp[0][0], mB[0][1], mG[0][1], mBe[0][1], bufB, H_SZ, 0);
  gemm_ln(bufB, H_SZ, wMlp[0][1], mB[0][2], mG[0][2], mBe[0][2], xcat, 2 * H_SZ, 0);
  // lo chain -> xcat[:, 512:1024]
  mlp_first_kernel<<<M_TOT, 256, 0, stream>>>(lo, mW[1][0], mB[1][0], mG[1][0], mBe[1][0],
                                              bufA, H_SZ, 0);
  gemm_ln(bufA, H_SZ, wMlp[1][0], mB[1][1], mG[1][1], mBe[1][1], bufB, H_SZ, 0);
  gemm_ln(bufB, H_SZ, wMlp[1][1], mB[1][2], mG[1][2], mBe[1][2], xcat, 2 * H_SZ, H_SZ);

  // gi_all = xcat @ Wih^T + bih  (N=1536 -> grid.y=3)
  gemm_kernel<4><<<dim3(MB, 3), 256, 0, stream>>>(
      xcat, 2 * H_SZ, 2 * H_SZ, wIh, bih, nullptr, nullptr,
      nullptr, 0, 0, gi, 3 * H_SZ, 3 * H_SZ, 0);

  // GRU recurrence -> bufA (bf16 hidden states)
  gru_kernel<<<1, 512, 0, stream>>>(gi, wHh, bhh, bufA);

  // fi chain
  gemm_ln(bufA, H_SZ, wFi[0], mB[2][0], mG[2][0], mBe[2][0], bufB, H_SZ, 0);
  gemm_ln(bufB, H_SZ, wFi[1], mB[2][1], mG[2][1], mBe[2][1], bufA, H_SZ, 0);
  gemm_ln(bufA, H_SZ, wFi[2], mB[2][2], mG[2][2], mBe[2][2], bufB, H_SZ, 0);

  // dense (N=64) -> raw logits
  gemm_kernel<1><<<dim3(MB, 1), 256, 0, stream>>>(
      bufB, H_SZ, H_SZ, wDen, denseB, nullptr, nullptr,
      nullptr, 0, 0, araw, P_SZ, P_SZ, 0);
  alpha_kernel<<<M_TOT, 64, 0, stream>>>(araw, alpha);

  // phase accumulation + synthesis
  phi_kernel<<<B_SZ, 256, 0, stream>>>(f0, phi);
  synth_kernel<<<(B_SZ * N_SMP + 255) / 256, 256, 0, stream>>>(lo, alpha, phi, y);
}
